// ElmanLSTM_32847909879808
// MI455X (gfx1250) — matrix operand order repaired and checked by reading, not compiled, not run on hardware
//
#include <hip/hip_runtime.h>
#include <math.h>

// ---------------------------------------------------------------------------
// LSTM on gfx1250 (MI455X): bf16 WMMA (f32 accum), fp32 cell state.
//   K1: init h double-buffers + barrier counter
//   K2: pack Wxh|Whh and Why into WMMA B-operand tile layout (bf16)
//   K3: pack x into WMMA A-operand tile layout (bf16), per timestep
//   K4: persistent 16-WG recurrent scan (weights LDS-resident, 240KB/WGP)
//   K5: output projection hs @ Why + by  (hs already A-packed by K4)
// ---------------------------------------------------------------------------

typedef __attribute__((ext_vector_type(16))) __bf16 v16bf;
typedef __attribute__((ext_vector_type(8)))  float  v8f;
typedef unsigned int U32;

#define SEQ     2048
#define BATCH   32
#define INPUT   256
#define HIDDEN  512
#define OUTPUT  256
#define GATES   2048          // 4*HIDDEN
#define NWG     16            // workgroups in the persistent scan kernel

// workspace byte offsets (tiles are 1 KiB = 32 lanes * 32 B)
#define OFF_WC   ((size_t)0)                           // 24*128 tiles = 3 MiB
#define OFF_WHY  (OFF_WC  + (size_t)24*128*1024)       // 16*16 tiles = 256 KiB
#define OFF_XP   (OFF_WHY + (size_t)16*16*1024)        // SEQ*16 tiles = 32 MiB
#define OFF_HSP  (OFF_XP  + (size_t)SEQ*16*1024)       // SEQ*32 tiles = 64 MiB
#define OFF_H0   (OFF_HSP + (size_t)SEQ*32*1024)       // 64 KiB (h, fp32)
#define OFF_H1   (OFF_H0  + (size_t)64*1024)           // 64 KiB
#define OFF_BAR  (OFF_H1  + (size_t)64*1024)           // barrier counter

// ---- helpers --------------------------------------------------------------

__device__ __forceinline__ unsigned short f32_to_bf16(float f) {
  const U32 u = __builtin_bit_cast(U32, f);
  return (unsigned short)((u + 0x7FFFu + ((u >> 16) & 1u)) >> 16);  // RNE
}
__device__ __forceinline__ U32 pack2_bf16(float a, float b) {
  return (U32)f32_to_bf16(a) | ((U32)f32_to_bf16(b) << 16);
}
__device__ __forceinline__ float sigmoidf_(float x) {
  return 1.f / (1.f + __expf(-x));
}

// software barrier across the 16 persistent workgroups (agent scope)
__device__ __forceinline__ void grid_barrier16(int* bar, int target) {
  __syncthreads();
  __threadfence();                                        // release dirty data
  if (threadIdx.x == 0) {
    __hip_atomic_fetch_add(bar, 1, __ATOMIC_RELEASE, __HIP_MEMORY_SCOPE_AGENT);
    while (__hip_atomic_load(bar, __ATOMIC_ACQUIRE, __HIP_MEMORY_SCOPE_AGENT) < target)
      __builtin_amdgcn_s_sleep(2);
  }
  __syncthreads();
  __threadfence();                                        // acquire (inv L0)
}

// ---- K1: init -------------------------------------------------------------

__global__ __launch_bounds__(256)
void ws_init_kernel(float* __restrict__ h0, float* __restrict__ h1,
                    int* __restrict__ bar) {
  const int stride = gridDim.x * blockDim.x;
  for (int i = blockIdx.x * blockDim.x + threadIdx.x; i < BATCH * HIDDEN; i += stride) {
    h0[i] = 0.f;
    h1[i] = 0.f;
  }
  if (blockIdx.x == 0 && threadIdx.x == 0) *bar = 0;
}

// ---- K2: pack weights into B-operand layout -------------------------------
// B tile (32K x 16N) bf16: lane l holds column N = l&15, K = (e&7) + ((l>>4)<<3) + ((e>>3)<<4),
// e = 0..15 packed 2/dword.  Tile = 1 KiB, lane stride 32 B.

__global__ __launch_bounds__(256)
void pack_weights_kernel(const float* __restrict__ Wxh, const float* __restrict__ Whh,
                         const float* __restrict__ Why,
                         U32* __restrict__ Wc, U32* __restrict__ WyP) {
  const int stride = gridDim.x * blockDim.x;
  const int NWc = 24 * 128 * 256;    // dwords: [kt=0..23][nt=0..127] tiles
  const int NWy = 16 * 16 * 256;     // dwords: [kt=0..15][nt=0..15]  tiles
  for (int d = blockIdx.x * blockDim.x + threadIdx.x; d < NWc + NWy; d += stride) {
    if (d < NWc) {
      const int tile = d >> 8, rem = d & 255;
      const int lane = rem >> 3, e2 = rem & 7;
      const int kt = tile >> 7, ntl = tile & 127;
      const int kr = ((e2 << 1) & 7) + ((lane >> 4) << 3) + ((e2 >> 2) << 4);
      const int n  = ntl * 16 + (lane & 15);
      const int k  = kt * 32 + kr;                 // combined K: [Wxh rows | Whh rows]
      const float* row = (k < INPUT) ? (Wxh + (size_t)k * GATES)
                                     : (Whh + (size_t)(k - INPUT) * GATES);
      Wc[d] = pack2_bf16(row[n], row[GATES + n]);  // k even -> k,k+1 same matrix
    } else {
      const int dd = d - NWc;
      const int tile = dd >> 8, rem = dd & 255;
      const int lane = rem >> 3, e2 = rem & 7;
      const int kt = tile >> 4, ntl = tile & 15;
      const int kr = ((e2 << 1) & 7) + ((lane >> 4) << 3) + ((e2 >> 2) << 4);
      const int n  = ntl * 16 + (lane & 15);
      const int k  = kt * 32 + kr;
      WyP[dd] = pack2_bf16(Why[(size_t)k * OUTPUT + n], Why[(size_t)(k + 1) * OUTPUT + n]);
    }
  }
}

// ---- K3: pack x into A-operand layout -------------------------------------
// A tile (16M x 32K) bf16: lane l holds row M = l&15,
// K = (e&7) + ((l>>4)<<3) + ((e>>3)<<4).  Per timestep: [mt=0,1][kt=0..7].

__global__ __launch_bounds__(256)
void pack_x_kernel(const float* __restrict__ x, U32* __restrict__ xP) {
  const size_t stride = (size_t)gridDim.x * blockDim.x;
  const size_t N = (size_t)SEQ * 16 * 256;  // dwords
  for (size_t d = (size_t)blockIdx.x * blockDim.x + threadIdx.x; d < N; d += stride) {
    const int tile = (int)(d >> 8), rem = (int)(d & 255);
    const int lane = rem >> 3, e2 = rem & 7;
    const int t  = tile >> 4;
    const int r  = tile & 15;
    const int mt = r >> 3, kt = r & 7;
    const int e  = e2 << 1;
    const int kr = (e & 7) + ((lane >> 4) << 3) + ((e >> 3) << 4);
    const int m  = mt * 16 + (lane & 15);
    const int k  = kt * 32 + kr;
    const float2 v = ((const float2*)x)[((size_t)m * (SEQ * INPUT) + (size_t)t * INPUT + k) >> 1];
    xP[d] = pack2_bf16(v.x, v.y);
  }
}

// ---- K4: persistent recurrent scan ----------------------------------------
// WG w owns hidden units j in [w*32, w*32+32): gate columns {j, j+512, j+1024,
// j+1536} => 8 N-tiles.  Per step: gates = [x_t|h] (32x768) @ Wc(:,slice),
// cell update in fp32, h slice exchanged through L2, agent barrier.
// LDS: 32 KiB packed h (A) + 192 KiB weights (B, loaded once) + 16 KiB gates.
// The reduction is split into a pure-global loop (x part) and a pure-LDS loop
// (h part), both fully unrolled: no generic/FLAT pointers, no branches on the
// serial critical path.

__global__ __launch_bounds__(256, 1)
void lstm_recur_kernel(const U32* __restrict__ Wc, const U32* __restrict__ xP,
                       U32* __restrict__ hsP,
                       float* __restrict__ hbuf0, float* __restrict__ hbuf1,
                       int* __restrict__ bar, const float* __restrict__ bh) {
  extern __shared__ char smem[];
  char*  hA = smem;                         //  32 KiB: h, A-packed bf16 [2mt][16kt]
  char*  Bc = smem + 32 * 1024;             // 192 KiB: weights [24kt][8nt] tiles
  float* gl = (float*)(smem + 224 * 1024);  //  16 KiB: gates [32 b][128 cols]

  const int w    = blockIdx.x;              // 0..15
  const int tid  = threadIdx.x;
  const int lane = tid & 31;
  const int wv   = tid >> 5;                // 8 wave32s

  // --- load this WG's 192 weight tiles into LDS (once) ---
  for (int i = tid; i < 192 * 64; i += 256) {
    const int tile = i >> 6;                // [kt][q*2+u]
    const int q4   = i & 63;
    const int kt   = tile >> 3;
    const int j    = tile & 7;
    const int ntg  = (j >> 1) * 32 + w * 2 + (j & 1);   // global N-tile
    ((float4*)(Bc + (size_t)tile * 1024))[q4] =
        ((const float4*)(Wc + (size_t)(kt * 128 + ntg) * 256))[q4];
  }

  // --- per-thread cell role: batch eb, hidden units jr0..jr0+3 ---
  const int eb  = tid >> 3;
  const int jr0 = (tid & 7) << 2;
  float bias[4][4];
#pragma unroll
  for (int q = 0; q < 4; ++q)
#pragma unroll
    for (int z = 0; z < 4; ++z)
      bias[q][z] = bh[q * HIDDEN + w * 32 + jr0 + z];
  float cst[4] = {0.f, 0.f, 0.f, 0.f};      // fp32 cell state, lives in VGPRs

  // --- wave -> two C tiles: mt = wv>>2, n-tile indices ni0, ni0+1 ---
  const int mt  = wv >> 2;
  const int ni0 = (wv & 3) * 2;
  const int ni1 = ni0 + 1;

  // per-wave invariant base pointers
  const char* bc0 = Bc + (size_t)ni0 * 1024 + lane * 32;  // + kt*8KiB
  const char* bc1 = Bc + (size_t)ni1 * 1024 + lane * 32;
  const char* hAb = hA + (size_t)mt * 16 * 1024 + lane * 32;  // + kt*1KiB

  __syncthreads();

  for (int t = 0; t < SEQ; ++t) {
    const float* __restrict__ hin  = (t & 1) ? hbuf1 : hbuf0;
    float* __restrict__       hout = (t & 1) ? hbuf0 : hbuf1;

    // 1. pack full h (fp32 32x512) -> LDS bf16 A-layout
    for (int i = tid; i < 8192; i += 256) {
      const int m  = i >> 8;
      const int kp = i & 255;
      const int k  = kp << 1;
      const float2 hv = ((const float2*)hin)[m * 256 + kp];
      const U32 pk = pack2_bf16(hv.x, hv.y);
      const int kt = k >> 5, kr = k & 31;
      const int g  = (kr >> 3) & 1;
      const int e  = (kr & 7) | ((kr >> 4) << 3);
      *(U32*)(hA + (size_t)(((m >> 4) * 16 + kt)) * 1024 + (g * 16 + (m & 15)) * 32 + e * 2) = pk;
    }
    __syncthreads();

    // 2a. GEMM, x part: 8 k-tiles streamed from packed x (global, L2-hot)
    v8f acc0 = {}; v8f acc1 = {};
    const char* xbase = (const char*)xP + (size_t)(t * 16 + mt * 8) * 1024 + lane * 32;
#pragma unroll
    for (int kt = 0; kt < 8; ++kt) {
      const v16bf a  = *(const v16bf*)(xbase + (size_t)kt * 1024);
      const v16bf b0 = *(const v16bf*)(bc0 + (size_t)kt * 8192);
      const v16bf b1 = *(const v16bf*)(bc1 + (size_t)kt * 8192);
      acc0 = __builtin_amdgcn_wmma_f32_16x16x32_bf16(false, a, false, b0, (short)0, acc0, false, false);
      acc1 = __builtin_amdgcn_wmma_f32_16x16x32_bf16(false, a, false, b1, (short)0, acc1, false, false);
    }
    // 2b. GEMM, h part: 16 k-tiles from LDS
#pragma unroll
    for (int kt = 0; kt < 16; ++kt) {
      const v16bf a  = *(const v16bf*)(hAb + (size_t)kt * 1024);
      const v16bf b0 = *(const v16bf*)(bc0 + (size_t)(kt + 8) * 8192);
      const v16bf b1 = *(const v16bf*)(bc1 + (size_t)(kt + 8) * 8192);
      acc0 = __builtin_amdgcn_wmma_f32_16x16x32_bf16(false, a, false, b0, (short)0, acc0, false, false);
      acc1 = __builtin_amdgcn_wmma_f32_16x16x32_bf16(false, a, false, b1, (short)0, acc1, false, false);
    }

    // 2c. prefetch next step's x tiles into caches while we do the cell work
#pragma unroll
    for (int kt = 0; kt < 8; ++kt)
      __builtin_prefetch(xbase + 16 * 1024 + (size_t)kt * 1024, 0, 1);

    // 3. dump C tiles to LDS gate buffer (col = ni*16 + n, row = batch)
    {
      const int nr    = lane & 15;
      const int mbase = mt * 16 + ((lane >> 4) << 3);
#pragma unroll
      for (int r = 0; r < 8; ++r) {
        gl[(mbase + r) * 128 + ni0 * 16 + nr] = acc0[r];
        gl[(mbase + r) * 128 + ni1 * 16 + nr] = acc1[r];
      }
    }
    __syncthreads();

    // 4. LSTM cell (fp32) for (eb, jr0..jr0+3)
    float hv4[4];
#pragma unroll
    for (int z = 0; z < 4; ++z) {
      const int jr = jr0 + z;
      const float ig = sigmoidf_(gl[eb * 128 +  0 + jr] + bias[0][z]);
      const float fg = sigmoidf_(gl[eb * 128 + 32 + jr] + bias[1][z]);
      const float gg = tanhf    (gl[eb * 128 + 64 + jr] + bias[2][z]);
      const float og = sigmoidf_(gl[eb * 128 + 96 + jr] + bias[3][z]);
      cst[z] = fg * cst[z] + ig * gg;
      hv4[z] = og * tanhf(cst[z]);
    }

    // h slice -> fp32 exchange buffer (read by all WGs next step)
    *(float4*)(hout + eb * HIDDEN + w * 32 + jr0) = make_float4(hv4[0], hv4[1], hv4[2], hv4[3]);

    // h slice -> hs, already bf16 A-packed for the output GEMM (k-tile == w)
    {
      const int g  = (jr0 >> 3) & 1;
      const int e0 = (jr0 & 7) | ((jr0 >> 4) << 3);
      const int ln = g * 16 + (eb & 15);
      const U32 lo = pack2_bf16(hv4[0], hv4[1]);
      const U32 hi = pack2_bf16(hv4[2], hv4[3]);
      *(uint2*)((char*)hsP + (size_t)((t * 2 + (eb >> 4)) * 16 + w) * 1024 + ln * 32 + e0 * 2) =
          make_uint2(lo, hi);
    }

    // 5. all-16-WG barrier: h exchange visible before next step
    grid_barrier16(bar, (t + 1) * NWG);
  }
}

// ---- K5: output projection out = hs @ Why + by ----------------------------
// hs rows are (t,b)-ordered: row = t*32 + b; 1 C-tile per wave, 16 WMMAs.

__global__ __launch_bounds__(256)
void out_proj_kernel(const U32* __restrict__ hsP, const U32* __restrict__ WyP,
                     const float* __restrict__ by, float* __restrict__ out) {
  const int lane = threadIdx.x & 31;
  const int wid  = blockIdx.x * 8 + (threadIdx.x >> 5);
  const int Mt   = wid >> 4;                // 0..4095
  const int nt   = wid & 15;
  const char* ab = (const char*)hsP + (size_t)Mt * 16 * 1024 + lane * 32;
  const char* bb = (const char*)WyP + (size_t)nt * 1024 + lane * 32;
  v8f acc = {};
#pragma unroll
  for (int kt = 0; kt < 16; ++kt) {
    if (kt < 15) __builtin_prefetch(ab + (size_t)(kt + 1) * 1024, 0, 1);
    const v16bf a = *(const v16bf*)(ab + (size_t)kt * 1024);
    const v16bf b = *(const v16bf*)(bb + (size_t)kt * 16 * 1024);
    acc = __builtin_amdgcn_wmma_f32_16x16x32_bf16(false, a, false, b, (short)0, acc, false, false);
  }
  const int n     = nt * 16 + (lane & 15);
  const float bia = by[n];
  const int rbase = Mt * 16 + ((lane >> 4) << 3);
#pragma unroll
  for (int r = 0; r < 8; ++r) {
    const int row = rbase + r;
    const int t = row >> 5;
    const int b = row & 31;
    out[(size_t)b * (SEQ * OUTPUT) + (size_t)t * OUTPUT + n] = acc[r] + bia;
  }
}

// ---- host entry -----------------------------------------------------------

extern "C" void kernel_launch(void* const* d_in, const int* in_sizes, int n_in,
                              void* d_out, int out_size, void* d_ws, size_t ws_size,
                              hipStream_t stream) {
  (void)in_sizes; (void)n_in; (void)out_size; (void)ws_size;
  const float* x   = (const float*)d_in[0];
  const float* Wxh = (const float*)d_in[1];
  const float* Whh = (const float*)d_in[2];
  const float* bh  = (const float*)d_in[3];
  const float* Why = (const float*)d_in[4];
  const float* by  = (const float*)d_in[5];
  float* out = (float*)d_out;

  char* ws   = (char*)d_ws;
  U32*   Wc  = (U32*)(ws + OFF_WC);
  U32*   WyP = (U32*)(ws + OFF_WHY);
  U32*   xP  = (U32*)(ws + OFF_XP);
  U32*   hsP = (U32*)(ws + OFF_HSP);
  float* h0  = (float*)(ws + OFF_H0);
  float* h1  = (float*)(ws + OFF_H1);
  int*   bar = (int*)(ws + OFF_BAR);

  // 240 KiB dynamic LDS (CDNA5 WGP has 320 KiB)
  (void)hipFuncSetAttribute((const void*)lstm_recur_kernel,
                            hipFuncAttributeMaxDynamicSharedMemorySize, 240 * 1024);

  ws_init_kernel     <<<64,   256, 0, stream>>>(h0, h1, bar);
  pack_weights_kernel<<<1024, 256, 0, stream>>>(Wxh, Whh, Why, Wc, WyP);
  pack_x_kernel      <<<8192, 256, 0, stream>>>(x, xP);
  lstm_recur_kernel  <<<NWG,  256, 240 * 1024, stream>>>(Wc, xP, hsP, h0, h1, bar, bh);
  out_proj_kernel    <<<8192, 256, 0, stream>>>(hsP, WyP, by, out);
}
